// SNN_10788957847678
// MI455X (gfx1250) — matrix operand order repaired and checked by reading, not compiled, not run on hardware
//
#include <hip/hip_runtime.h>

typedef __attribute__((ext_vector_type(16))) _Float16 v16h;
typedef __attribute__((ext_vector_type(8)))  _Float16 v8h;
typedef __attribute__((ext_vector_type(8)))  float    v8f;
typedef __attribute__((ext_vector_type(4)))  unsigned int v4u;
typedef __attribute__((ext_vector_type(8)))  int v8i;
typedef __attribute__((ext_vector_type(4)))  int v4i;

#define LIF_BETA 0.90483741803595957f

constexpr int Bn = 256;   // batch
constexpr int In = 784;   // input
constexpr int Hn = 1024;  // hidden
constexpr int On = 10;    // output
constexpr int Tn = 100;   // time steps
constexpr int BH = Bn * Hn;

// ---------------- small utility kernels ----------------

__global__ void k_zero_f32(float* __restrict__ p, int n) {
    int i = blockIdx.x * blockDim.x + threadIdx.x;
    if (i < n) p[i] = 0.0f;
}

__global__ void k_zero_f16(_Float16* __restrict__ p, int n) {
    int i = blockIdx.x * blockDim.x + threadIdx.x;
    if (i < n) p[i] = (_Float16)0.0f;
}

// in_cur = x @ Wf1, full f32 precision (once per call)
__global__ __launch_bounds__(256) void k_incur(const float* __restrict__ x,
                                               const float* __restrict__ wf1,
                                               float* __restrict__ out) {
    int i = blockIdx.x * 256 + threadIdx.x;   // Bn*Hn threads
    int h = i & (Hn - 1);
    int b = i >> 10;
    const float* xr = x + (size_t)b * In;
    float acc = 0.0f;
    for (int k = 0; k < In; ++k)
        acc = fmaf(xr[k], wf1[(size_t)k * Hn + h], acc);
    out[i] = acc;
}

// Swizzle a [Hn][Hn] f32 row-major weight into fragment-ready f16 layout:
// o[(((nt*32)+kt)*32 + L)*16 + j] = W[k][n],  n = nt*16+(L&15),  k = kt*32+(j>>3)*16+(L>>4)*8+(j&7)
__global__ void k_swz_w(const float* __restrict__ w, _Float16* __restrict__ o) {
    int i = blockIdx.x * 256 + threadIdx.x;   // Hn*Hn = 1<<20
    int j  = i & 15;
    int L  = (i >> 4) & 31;
    int kt = (i >> 9) & 31;
    int nt = i >> 14;                          // 0..63
    int n = nt * 16 + (L & 15);
    int k = kt * 32 + ((j >> 3) << 4) + ((L >> 4) << 3) + (j & 7);
    o[i] = (_Float16)w[(size_t)k * Hn + n];
}

// Same swizzle for Wro [Hn][On], N padded to 16 with zeros. Output 32*32*16 elems.
__global__ void k_swz_wro(const float* __restrict__ w, _Float16* __restrict__ o) {
    int i = blockIdx.x * 256 + threadIdx.x;
    if (i >= 32 * 32 * 16) return;
    int j  = i & 15;
    int L  = (i >> 4) & 31;
    int kt = i >> 9;                           // 0..31
    int n = L & 15;
    int k = kt * 32 + ((j >> 3) << 4) + ((L >> 4) << 3) + (j & 7);
    o[i] = (_Float16)((n < On) ? w[(size_t)k * On + n] : 0.0f);
}

// ---------------- fragment helpers ----------------

__device__ inline v16h cat16(v8h lo, v8h hi) {
    v16h r;
#pragma unroll
    for (int j = 0; j < 8; ++j) { r[j] = lo[j]; r[j + 8] = hi[j]; }
    return r;
}

// A-fragment from LDS tile As[row][32]: lanes 0-15 K{0-7,16-23}, lanes 16-31 K{8-15,24-31}
__device__ inline v16h a_frag(const _Float16* As, int rowBase, int lane) {
    int row = rowBase + (lane & 15);
    int kb  = (lane >> 4) * 8;
    v8h lo = *(const v8h*)(As + row * 32 + kb);
    v8h hi = *(const v8h*)(As + row * 32 + kb + 16);
    return cat16(lo, hi);
}

// B-fragment from swizzled weights: one coalesced 32B chunk per lane
__device__ inline v16h b_frag(const _Float16* wswz, int nt, int kt, int lane) {
    const _Float16* p = wswz + (((size_t)nt * 32 + kt) * 32 + lane) * 16;
    v8h lo = *(const v8h*)p;
    v8h hi = *(const v8h*)(p + 8);
    return cat16(lo, hi);
}

// ---------------- TDM: DMA a 64x32 f16 tile (row stride Hn) into packed LDS ----------
#if __has_builtin(__builtin_amdgcn_tensor_load_to_lds)
#define USE_TDM 1
__device__ inline void tdm_load_tile_64x32(unsigned ldsAddr, const _Float16* gsrc) {
    unsigned long long ga = (unsigned long long)(uintptr_t)gsrc;
    v4u g0;
    g0[0] = 1u;                                           // count=1, user mode
    g0[1] = ldsAddr;                                      // lds_addr (bytes)
    g0[2] = (unsigned)(ga & 0xFFFFFFFFull);               // global_addr[31:0]
    g0[3] = (unsigned)((ga >> 32) & 0x01FFFFFFull)        // global_addr[56:32]
            | 0x80000000u;                                // type = 2 ("image")
    v8i g1;
    g1[0] = 0x00010000;            // data_size = 1 (2 bytes)
    g1[1] = (int)(1024u << 16);    // tensor_dim0 = 1024 (elements per row)
    g1[2] = (int)(256u << 16);     // tensor_dim1 = 256 (rows)
    g1[3] = (int)(32u << 16);      // tile_dim0 = 32
    g1[4] = 64;                    // tile_dim1 = 64, tile_dim2 = 0
    g1[5] = 1024;                  // tensor_dim0_stride = 1024
    g1[6] = 0;
    g1[7] = 0;
    v4i z4 = {0, 0, 0, 0};         // groups 2/3 unused (2-D tensor)
    v8i z8 = {0, 0, 0, 0, 0, 0, 0, 0};
    __builtin_amdgcn_tensor_load_to_lds(g0, g1, z4, z4, z8, 0);
}
#else
#define USE_TDM 0
#endif

// ---------------- fused GEMM + LIF layer ----------------
// block = 128 threads (4 waves, 2M x 2N), block tile 64(M) x 128(N),
// wave tile 32(M) x 64(N): acc[2][4], 8 WMMAs per 32-K step.
// A tile staged into LDS by the Tensor Data Mover (double buffered, overlapped
// with WMMA); B read directly from swizzled global (coalesced, L2-resident).
template <bool HAS_INCUR, int PASSES>
__global__ __launch_bounds__(128) void k_lif(
    const _Float16* __restrict__ A0, const _Float16* __restrict__ W0,
    const _Float16* __restrict__ A1, const _Float16* __restrict__ W1,
    const float* __restrict__ inCur,
    float* __restrict__ V,
    const _Float16* __restrict__ Zprev,
    _Float16* __restrict__ Znext) {
    __shared__ __align__(16) _Float16 As[2][64 * 32];

    int tid  = threadIdx.x;
    int lane = tid & 31;
    int wave = tid >> 5;          // 0..3
    int wm = wave >> 1;           // 0..1
    int wn = wave & 1;            // 0..1
    int m0 = blockIdx.y * 64;     // block M base
    int n0 = blockIdx.x * 128;    // block N base
    int mw = wm * 32;             // wave M offset inside block
    int nw = n0 + wn * 64;        // wave N base (global)
    int ntBase = nw >> 4;         // first 16-wide N fragment index
    bool tdmWave = (wave == 0);

    v8f acc[2][4] = {};

#if USE_TDM
    unsigned ldsA[2] = {(unsigned)(uintptr_t)&As[0][0], (unsigned)(uintptr_t)&As[1][0]};
#else
    int g0i = tid * 2;
    int sRow0 = g0i >> 2,       sKg0 = (g0i & 3) * 8;
    int sRow1 = (g0i + 1) >> 2, sKg1 = ((g0i + 1) & 3) * 8;
#endif

    for (int pass = 0; pass < PASSES; ++pass) {
        const _Float16* Ag = pass ? A1 : A0;
        const _Float16* Wg = pass ? W1 : W0;
        // prime buffer 0 with K tile 0
#if USE_TDM
        if (tdmWave) {
            tdm_load_tile_64x32(ldsA[0], Ag + (size_t)m0 * Hn);
            __builtin_amdgcn_s_wait_tensorcnt(0);
        }
#else
        *(v8h*)(&As[0][sRow0 * 32 + sKg0]) =
            *(const v8h*)(Ag + (size_t)(m0 + sRow0) * Hn + sKg0);
        *(v8h*)(&As[0][sRow1 * 32 + sKg1]) =
            *(const v8h*)(Ag + (size_t)(m0 + sRow1) * Hn + sKg1);
#endif
        __syncthreads();

        for (int kt = 0; kt < 32; ++kt) {
            int p = kt & 1;
            if (kt + 1 < 32) {   // stage next K tile into the other buffer
#if USE_TDM
                if (tdmWave)
                    tdm_load_tile_64x32(ldsA[p ^ 1],
                                        Ag + (size_t)m0 * Hn + (kt + 1) * 32);
#else
                int kb = (kt + 1) * 32;
                *(v8h*)(&As[p ^ 1][sRow0 * 32 + sKg0]) =
                    *(const v8h*)(Ag + (size_t)(m0 + sRow0) * Hn + kb + sKg0);
                *(v8h*)(&As[p ^ 1][sRow1 * 32 + sKg1]) =
                    *(const v8h*)(Ag + (size_t)(m0 + sRow1) * Hn + kb + sKg1);
#endif
            }
            v16h af[2], bf[4];
#pragma unroll
            for (int mi = 0; mi < 2; ++mi) af[mi] = a_frag(As[p], mw + mi * 16, lane);
#pragma unroll
            for (int ni = 0; ni < 4; ++ni) bf[ni] = b_frag(Wg, ntBase + ni, kt, lane);
#pragma unroll
            for (int mi = 0; mi < 2; ++mi)
#pragma unroll
                for (int ni = 0; ni < 4; ++ni)
                    acc[mi][ni] = __builtin_amdgcn_wmma_f32_16x16x32_f16(
                        false, af[mi], false, bf[ni], (short)0, acc[mi][ni], false, false);
#if USE_TDM
            if (tdmWave) __builtin_amdgcn_s_wait_tensorcnt(0);
#endif
            __syncthreads();
        }
    }

    // LIF epilogue: c[r] -> (M = r + 8*(lane>=16), N = lane&15)
    int colOff = lane & 15;
    int rowOff = (lane >> 4) * 8;
#pragma unroll
    for (int mi = 0; mi < 2; ++mi) {
#pragma unroll
        for (int ni = 0; ni < 4; ++ni) {
            int n = nw + ni * 16 + colOff;
#pragma unroll
            for (int r = 0; r < 8; ++r) {
                int m = m0 + mw + mi * 16 + rowOff + r;
                size_t idx = (size_t)m * Hn + n;
                float cur = acc[mi][ni][r];
                if (HAS_INCUR) cur += inCur[idx];
                float zp = (float)Zprev[idx];
                float v  = (LIF_BETA * V[idx] + cur) * (1.0f - zp);
                V[idx] = v;
                Znext[idx] = (_Float16)((v > 1.0f) ? 1.0f : 0.0f);
            }
        }
    }
}

// ---------------- fused-path readout: all R(t) = Z2(t)@Wro in parallel ----------------
__global__ __launch_bounds__(256) void k_readout_R(
    const _Float16* __restrict__ Z2seq,   // [Tn+1][BH], slot t+1 = spikes of step t
    const _Float16* __restrict__ WroSwz,
    float* __restrict__ R) {              // [b][t][16]
    int tid = threadIdx.x, lane = tid & 31, wave = tid >> 5;
    int gw = blockIdx.x * 8 + wave;
    int t  = gw >> 4;
    int mt = gw & 15;
    const _Float16* Z = Z2seq + (size_t)(t + 1) * BH;
    int arow = mt * 16 + (lane & 15);
    int akb  = (lane >> 4) * 8;
    v8f acc = {};
    for (int kt = 0; kt < 32; ++kt) {
        v8h alo = *(const v8h*)(Z + (size_t)arow * Hn + kt * 32 + akb);
        v8h ahi = *(const v8h*)(Z + (size_t)arow * Hn + kt * 32 + akb + 16);
        v16h a = cat16(alo, ahi);
        v16h b = b_frag(WroSwz, 0, kt, lane);
        acc = __builtin_amdgcn_wmma_f32_16x16x32_f16(false, a, false, b, (short)0, acc,
                                                     false, false);
    }
    int n = lane & 15;
#pragma unroll
    for (int r = 0; r < 8; ++r) {
        int m = mt * 16 + (lane >> 4) * 8 + r;
        R[((size_t)m * Tn + t) * 16 + n] = acc[r];
    }
}

// scan over t: Vr(t) = BETA*Vr(t-1) + R(t) + bias; write trace
__global__ void k_scan(const float* __restrict__ R, const float* __restrict__ bias,
                       float* __restrict__ out) {
    int i = blockIdx.x * blockDim.x + threadIdx.x;   // Bn*On
    if (i >= Bn * On) return;
    int b = i / On, n = i % On;
    float bb = bias[n];
    float v = 0.0f;
    const float* r = R + (size_t)b * Tn * 16 + n;
    for (int t = 0; t < Tn; ++t) {
        v = LIF_BETA * v + r[(size_t)t * 16] + bb;
        out[(size_t)b * Tn * On + t * On + n] = v;
    }
}

// ---------------- fallback per-step readout (small workspace) ----------------
__global__ __launch_bounds__(32) void k_readout_step(
    const _Float16* __restrict__ Z2, const _Float16* __restrict__ WroSwz,
    const float* __restrict__ bias, float* __restrict__ Vr,
    float* __restrict__ out, int t) {
    int lane = threadIdx.x;
    int mt = blockIdx.x;
    int arow = mt * 16 + (lane & 15);
    int akb  = (lane >> 4) * 8;
    v8f acc = {};
    for (int kt = 0; kt < 32; ++kt) {
        v8h alo = *(const v8h*)(Z2 + (size_t)arow * Hn + kt * 32 + akb);
        v8h ahi = *(const v8h*)(Z2 + (size_t)arow * Hn + kt * 32 + akb + 16);
        v16h a = cat16(alo, ahi);
        v16h b = b_frag(WroSwz, 0, kt, lane);
        acc = __builtin_amdgcn_wmma_f32_16x16x32_f16(false, a, false, b, (short)0, acc,
                                                     false, false);
    }
    int n = lane & 15;
    if (n < On) {
#pragma unroll
        for (int r = 0; r < 8; ++r) {
            int m = mt * 16 + (lane >> 4) * 8 + r;
            float v = LIF_BETA * Vr[m * On + n] + acc[r] + bias[n];
            Vr[m * On + n] = v;
            out[(size_t)m * Tn * On + t * On + n] = v;
        }
    }
}

// ---------------- host side ----------------

extern "C" void kernel_launch(void* const* d_in, const int* in_sizes, int n_in,
                              void* d_out, int out_size, void* d_ws, size_t ws_size,
                              hipStream_t stream) {
    const float* x    = (const float*)d_in[0];
    const float* Wf1  = (const float*)d_in[1];
    const float* Wr1  = (const float*)d_in[2];
    const float* Wf2  = (const float*)d_in[3];
    const float* Wr2  = (const float*)d_in[4];
    const float* Wro  = (const float*)d_in[5];
    const float* bias = (const float*)d_in[6];
    float* out = (float*)d_out;

    char* wsp = (char*)d_ws;
    size_t used = 0;
    auto carve = [&](size_t bytes) -> void* {
        void* p = (void*)(wsp + used);
        used += (bytes + 255) & ~(size_t)255;
        return p;
    };
    float*    inCur  = (float*)carve((size_t)BH * 4);
    float*    V1     = (float*)carve((size_t)BH * 4);
    float*    V2     = (float*)carve((size_t)BH * 4);
    _Float16* Z1[2]  = {(_Float16*)carve((size_t)BH * 2), (_Float16*)carve((size_t)BH * 2)};
    _Float16* Wr1s   = (_Float16*)carve((size_t)Hn * Hn * 2);
    _Float16* Wf2s   = (_Float16*)carve((size_t)Hn * Hn * 2);
    _Float16* Wr2s   = (_Float16*)carve((size_t)Hn * Hn * 2);
    _Float16* WroS   = (_Float16*)carve((size_t)32 * 32 * 16 * 2);
    size_t baseUsed = used;

    size_t seqBytes = (size_t)(Tn + 1) * BH * 2;
    size_t rBytes   = (size_t)Bn * Tn * 16 * 4;
    bool fused = (baseUsed + ((seqBytes + 255) & ~(size_t)255) +
                  ((rBytes + 255) & ~(size_t)255)) <= ws_size;

    k_swz_w<<<(Hn * Hn) / 256, 256, 0, stream>>>(Wr1, Wr1s);
    k_swz_w<<<(Hn * Hn) / 256, 256, 0, stream>>>(Wf2, Wf2s);
    k_swz_w<<<(Hn * Hn) / 256, 256, 0, stream>>>(Wr2, Wr2s);
    k_swz_wro<<<(32 * 32 * 16 + 255) / 256, 256, 0, stream>>>(Wro, WroS);
    k_zero_f32<<<(BH + 255) / 256, 256, 0, stream>>>(V1, BH);
    k_zero_f32<<<(BH + 255) / 256, 256, 0, stream>>>(V2, BH);
    k_zero_f16<<<(BH + 255) / 256, 256, 0, stream>>>(Z1[0], BH);
    k_incur<<<BH / 256, 256, 0, stream>>>(x, Wf1, inCur);

    dim3 grid(Hn / 128, Bn / 64);   // 8 x 4 blocks, 128 threads

    if (fused) {
        _Float16* Z2seq = (_Float16*)carve(seqBytes);
        float*    R     = (float*)carve(rBytes);
        k_zero_f16<<<(BH + 255) / 256, 256, 0, stream>>>(Z2seq, BH);  // slot 0

        for (int t = 0; t < Tn; ++t) {
            int cur = t & 1, nxt = cur ^ 1;
            k_lif<true, 1><<<grid, 128, 0, stream>>>(
                Z1[cur], Wr1s, nullptr, nullptr, inCur, V1, Z1[cur], Z1[nxt]);
            k_lif<false, 2><<<grid, 128, 0, stream>>>(
                Z1[nxt], Wf2s, Z2seq + (size_t)t * BH, Wr2s, nullptr, V2,
                Z2seq + (size_t)t * BH, Z2seq + (size_t)(t + 1) * BH);
        }
        k_readout_R<<<(Tn * 16) / 8, 256, 0, stream>>>(Z2seq, WroS, R);
        k_scan<<<(Bn * On + 255) / 256, 256, 0, stream>>>(R, bias, out);
    } else {
        _Float16* Z2[2] = {(_Float16*)carve((size_t)BH * 2),
                           (_Float16*)carve((size_t)BH * 2)};
        float* Vr = (float*)carve((size_t)Bn * On * 4);
        k_zero_f16<<<(BH + 255) / 256, 256, 0, stream>>>(Z2[0], BH);
        k_zero_f32<<<(Bn * On + 255) / 256, 256, 0, stream>>>(Vr, Bn * On);

        for (int t = 0; t < Tn; ++t) {
            int cur = t & 1, nxt = cur ^ 1;
            k_lif<true, 1><<<grid, 128, 0, stream>>>(
                Z1[cur], Wr1s, nullptr, nullptr, inCur, V1, Z1[cur], Z1[nxt]);
            k_lif<false, 2><<<grid, 128, 0, stream>>>(
                Z1[nxt], Wf2s, Z2[cur], Wr2s, nullptr, V2, Z2[cur], Z2[nxt]);
            k_readout_step<<<16, 32, 0, stream>>>(Z2[nxt], WroS, bias, Vr, out, t);
        }
    }
    (void)in_sizes; (void)n_in; (void)out_size;
}
